// PVMLayer3_11639361372336
// MI455X (gfx1250) — compile-verified, matrix operand order checked by reading, not run on hardware
//
#include <hip/hip_runtime.h>
#include <hip/hip_bf16.h>
#include <math.h>

#define BATCH   4
#define INDIM   192
#define LSEQ    4096
#define DMODEL  64
#define DSTATE  16
#define DINNER  512
#define DTRANK  4
#define NSEQ    12        // 3 * BATCH
#define XPPAD   48        // DT_RANK + 2*D_STATE (36) padded to 3 WMMA tiles
#define LN_EPS  1e-5f

typedef __attribute__((ext_vector_type(16))) __bf16 v16bf;
typedef __attribute__((ext_vector_type(8)))  __bf16 v8bf;
typedef __attribute__((ext_vector_type(8)))  float  v8f;

// ---------------------------------------------------------------------------
// WMMA helpers (gfx1250, wave32). 16-bit A/B fragment layout per CDNA5 ISA:
// lane<16 holds row (lane&15) K-halves {0..7,16..23}; lane>=16 holds {8..15,24..31}.
// ---------------------------------------------------------------------------
__device__ __forceinline__ v16bf load_frag(const __bf16* tile, int ld) {
  const int lane = threadIdx.x & 31;
  const __bf16* p = tile + (size_t)(lane & 15) * ld + ((lane >> 4) << 3);
  v8bf lo = *(const v8bf*)(p);
  v8bf hi = *(const v8bf*)(p + 16);
  v16bf f;
#pragma unroll
  for (int e = 0; e < 8; ++e) { f[e] = lo[e]; f[e + 8] = hi[e]; }
  return f;
}

__device__ __forceinline__ v8f wmma_bf16(v16bf a, v16bf b, v8f c) {
  return __builtin_amdgcn_wmma_f32_16x16x32_bf16(false, a, false, b, (short)0, c,
                                                 false, false);
}

// One 16-row strip x (NT*16) cols, K-loop in steps of 32. A-frag reused NT times.
template <int NT>
__device__ __forceinline__ void gemm_acc(const __bf16* A, int lda,
                                         const __bf16* W, int ldw, int K, v8f* acc) {
  for (int k = 0; k < K; k += 32) {
    v16bf a = load_frag(A + k, lda);
#pragma unroll
    for (int j = 0; j < NT; ++j) {
      v16bf b = load_frag(W + (size_t)(j * 16) * ldw + k, ldw);
      acc[j] = wmma_bf16(a, b, acc[j]);
    }
  }
}

// ---------------------------------------------------------------------------
// K0: convert all weight matrices to bf16 (x_proj zero-padded 36 -> 48 rows)
// ---------------------------------------------------------------------------
__global__ void k_prep_weights(const float* Win, const float* Wxp, const float* Wout,
                               const float* Wpj, __bf16* bWin, __bf16* bWxp,
                               __bf16* bWout, __bf16* bWpj) {
  int i = blockIdx.x * blockDim.x + threadIdx.x;
  const int n_in = 1024 * 64, n_xp = XPPAD * 512, n_out = 64 * 512, n_pj = 192 * 192;
  if (i < n_in) { bWin[i] = (__bf16)Win[i]; return; }
  i -= n_in;
  if (i < n_xp) { bWxp[i] = (i < 36 * 512) ? (__bf16)Wxp[i] : (__bf16)0.0f; return; }
  i -= n_xp;
  if (i < n_out) { bWout[i] = (__bf16)Wout[i]; return; }
  i -= n_out;
  if (i < n_pj) { bWpj[i] = (__bf16)Wpj[i]; }
}

// ---------------------------------------------------------------------------
// K1: LayerNorm over 192 channels of NCHW input + split into xs (12,4096,64)
// thread = one (b,l) position; channel loads coalesced across l.
// ---------------------------------------------------------------------------
__global__ void k_ln1(const float* __restrict__ x, const float* __restrict__ g,
                      const float* __restrict__ be, float* xs, __bf16* xsb) {
  int idx = blockIdx.x * blockDim.x + threadIdx.x;   // 16384
  int b = idx >> 12, l = idx & 4095;
  const float* px = x + (size_t)b * INDIM * LSEQ + l;
  float s = 0.f, s2 = 0.f;
  for (int c = 0; c < INDIM; ++c) {
    float v = px[(size_t)c * LSEQ];
    s += v; s2 += v * v;
  }
  float mu = s * (1.f / INDIM);
  float var = s2 * (1.f / INDIM) - mu * mu;
  float rstd = rsqrtf(var + LN_EPS);
  for (int c = 0; c < INDIM; ++c) {
    float v = (px[(size_t)c * LSEQ] - mu) * rstd * g[c] + be[c];
    int i3 = c >> 6, dm = c & 63;
    int nsq = i3 * BATCH + b;                        // stack order: n = i*B + b
    size_t o = ((size_t)nsq * LSEQ + l) * DMODEL + dm;
    xs[o] = v; xsb[o] = (__bf16)v;
  }
}

// ---------------------------------------------------------------------------
// K2: in_proj  xz[49152,1024] = xs[49152,64] @ Win^T   (WMMA bf16)
// ---------------------------------------------------------------------------
__global__ void k_gemm_inproj(const __bf16* __restrict__ xsb,
                              const __bf16* __restrict__ bWin, float* xz) {
  const int NT = 4, NG = 1024 / (16 * NT);           // 16 column groups
  int wave = blockIdx.x * (blockDim.x >> 5) + (threadIdx.x >> 5);
  int tm = wave / NG, tg = wave % NG;
  const __bf16* A = xsb + (size_t)tm * 16 * 64;
  const __bf16* W = bWin + (size_t)tg * (16 * NT) * 64;
  v8f acc[NT] = {};
  gemm_acc<NT>(A, 64, W, 64, 64, acc);
  int lane = threadIdx.x & 31, nn = lane & 15, mb = (lane >> 4) << 3;
  float* C = xz + (size_t)tm * 16 * 1024 + tg * (16 * NT);
#pragma unroll
  for (int j = 0; j < NT; ++j)
#pragma unroll
    for (int r = 0; r < 8; ++r)
      C[(size_t)(mb + r) * 1024 + j * 16 + nn] = acc[j][r];
}

// ---------------------------------------------------------------------------
// K3: depthwise causal conv (k=4) + SiLU.  xi = xz[...,:512]
// ---------------------------------------------------------------------------
__global__ void k_conv(const float* __restrict__ xz, const float* __restrict__ cW,
                       const float* __restrict__ cb, float* xc, __bf16* xcb) {
  size_t idx = (size_t)blockIdx.x * blockDim.x + threadIdx.x;  // 25165824
  int d = (int)(idx & 511);
  size_t row = idx >> 9;                     // n*L + t
  int t = (int)(row & 4095);
  float acc = cb[d];
  const float* w = cW + d * 4;
#pragma unroll
  for (int k = 0; k < 4; ++k) {
    int tt = t - 3 + k;
    if (tt >= 0) acc += w[k] * xz[(row - (size_t)(t - tt)) * 1024 + d];
  }
  float sv = acc / (1.f + __expf(-acc));     // silu
  xc[idx] = sv; xcb[idx] = (__bf16)sv;
}

// ---------------------------------------------------------------------------
// K4: x_proj  dbl[49152,48] = xc[49152,512] @ Wxp_pad^T   (WMMA bf16, NT=3)
// ---------------------------------------------------------------------------
__global__ void k_gemm_xproj(const __bf16* __restrict__ xcb,
                             const __bf16* __restrict__ bWxp, float* dbl) {
  const int NT = 3;
  int wave = blockIdx.x * (blockDim.x >> 5) + (threadIdx.x >> 5); // 0..3071
  const __bf16* A = xcb + (size_t)wave * 16 * 512;
  v8f acc[NT] = {};
  gemm_acc<NT>(A, 512, bWxp, 512, 512, acc);
  int lane = threadIdx.x & 31, nn = lane & 15, mb = (lane >> 4) << 3;
  float* C = dbl + (size_t)wave * 16 * XPPAD;
#pragma unroll
  for (int j = 0; j < NT; ++j)
#pragma unroll
    for (int r = 0; r < 8; ++r)
      C[(size_t)(mb + r) * XPPAD + j * 16 + nn] = acc[j][r];
}

// ---------------------------------------------------------------------------
// K5: dt = softplus(dbl[...,:4] @ dtW^T + dtb)   (K=4 -> plain VALU kernel)
// ---------------------------------------------------------------------------
__global__ void k_dt(const float* __restrict__ dbl, const float* __restrict__ dtW,
                     const float* __restrict__ dtb, float* dtf) {
  size_t idx = (size_t)blockIdx.x * blockDim.x + threadIdx.x;  // 25165824
  int d = (int)(idx & 511);
  size_t row = idx >> 9;
  const float* p = dbl + row * XPPAD;
  float a = dtb[d];
#pragma unroll
  for (int r = 0; r < DTRANK; ++r) a += p[r] * dtW[d * 4 + r];
  dtf[idx] = (a > 20.f) ? a : log1pf(__expf(a));
}

// ---------------------------------------------------------------------------
// K6: selective scan, fused with +x*D and SiLU(z) gate. One lane per (n,d);
// 16-state recurrence in VGPRs; B/C staged through LDS in 64-step chunks.
// ---------------------------------------------------------------------------
#define TCH 64
__global__ void k_scan(const float* __restrict__ dbl, const float* __restrict__ dtf,
                       const float* __restrict__ xc, const float* __restrict__ xz,
                       const float* __restrict__ Alog, const float* __restrict__ Dp,
                       __bf16* yb) {
  __shared__ float lB[TCH][DSTATE];
  __shared__ float lC[TCH][DSTATE];
  int n = blockIdx.x >> 1;
  int d = ((blockIdx.x & 1) << 8) + threadIdx.x;
  float a_s[DSTATE], h[DSTATE];
#pragma unroll
  for (int s = 0; s < DSTATE; ++s) {
    a_s[s] = -__expf(Alog[d * DSTATE + s]);
    h[s] = 0.f;
  }
  float Dd = Dp[d];
  for (int tc = 0; tc < LSEQ; tc += TCH) {
    for (int j = threadIdx.x; j < TCH * 32; j += 256) {
      int st = j >> 5, c = j & 31;
      float v = dbl[((size_t)n * LSEQ + tc + st) * XPPAD + DTRANK + c];
      if (c < DSTATE) lB[st][c] = v; else lC[st][c - DSTATE] = v;
    }
    __syncthreads();
    for (int s = 0; s < TCH; ++s) {
      size_t row = (size_t)n * LSEQ + tc + s;
      float dt = dtf[row * DINNER + d];
      float xv = xc[row * DINNER + d];
      float dtx = dt * xv;
      float y = 0.f;
#pragma unroll
      for (int st = 0; st < DSTATE; ++st) {
        h[st] = h[st] * __expf(dt * a_s[st]) + dtx * lB[s][st];
        y += h[st] * lC[s][st];
      }
      y += xv * Dd;
      float z = xz[row * 1024 + DINNER + d];
      y *= z / (1.f + __expf(-z));            // gate: y * silu(z)
      yb[row * DINNER + d] = (__bf16)y;
    }
    __syncthreads();
  }
}

// ---------------------------------------------------------------------------
// K7: out_proj GEMM (49152x512x64) + skip add + recombine to (B,L,192)
// ---------------------------------------------------------------------------
__global__ void k_gemm_outproj(const __bf16* __restrict__ yb,
                               const __bf16* __restrict__ bWout,
                               const float* __restrict__ xs,
                               const float* __restrict__ skip, float* ycat) {
  const int NT = 4;                                  // N = 64
  int wave = blockIdx.x * (blockDim.x >> 5) + (threadIdx.x >> 5); // 0..3071
  const __bf16* A = yb + (size_t)wave * 16 * 512;
  v8f acc[NT] = {};
  gemm_acc<NT>(A, 512, bWout, 512, 512, acc);
  float sk = skip[0];
  int lane = threadIdx.x & 31, nn = lane & 15, mb = (lane >> 4) << 3;
#pragma unroll
  for (int j = 0; j < NT; ++j) {
    int dm = j * 16 + nn;
#pragma unroll
    for (int r = 0; r < 8; ++r) {
      int m = wave * 16 + mb + r;                    // m = nsq*L + t
      int nsq = m >> 12, t = m & 4095;
      int i3 = nsq >> 2, b = nsq & 3;                // nsq = i*4 + b
      float v = acc[j][r] + sk * xs[(size_t)m * DMODEL + dm];
      ycat[(((size_t)b << 12) + t) * INDIM + (i3 << 6) + dm] = v;
    }
  }
}

// ---------------------------------------------------------------------------
// K8: second LayerNorm over 192 (row-contiguous), emit bf16
// ---------------------------------------------------------------------------
__global__ void k_ln2(const float* __restrict__ ycat, const float* __restrict__ g,
                      const float* __restrict__ be, __bf16* ylnb) {
  int idx = blockIdx.x * blockDim.x + threadIdx.x;   // 16384
  const float* p = ycat + (size_t)idx * INDIM;
  float s = 0.f, s2 = 0.f;
  for (int c = 0; c < INDIM; ++c) { float v = p[c]; s += v; s2 += v * v; }
  float mu = s * (1.f / INDIM);
  float var = s2 * (1.f / INDIM) - mu * mu;
  float rstd = rsqrtf(var + LN_EPS);
  __bf16* q = ylnb + (size_t)idx * INDIM;
  for (int c = 0; c < INDIM; ++c)
    q[c] = (__bf16)((p[c] - mu) * rstd * g[c] + be[c]);
}

// ---------------------------------------------------------------------------
// K9: final projection GEMM (16384x192x192) + bias, write NCHW-transposed
// ---------------------------------------------------------------------------
__global__ void k_gemm_proj(const __bf16* __restrict__ ylnb,
                            const __bf16* __restrict__ bWpj,
                            const float* __restrict__ pb, float* out) {
  const int NT = 4, NG = 192 / (16 * NT);            // 3 column groups
  int wave = blockIdx.x * (blockDim.x >> 5) + (threadIdx.x >> 5);
  int tm = wave / NG, tg = wave % NG;
  const __bf16* A = ylnb + (size_t)tm * 16 * INDIM;
  const __bf16* W = bWpj + (size_t)tg * 64 * INDIM;
  v8f acc[NT] = {};
  gemm_acc<NT>(A, INDIM, W, INDIM, INDIM, acc);
  int lane = threadIdx.x & 31, nn = lane & 15, mb = (lane >> 4) << 3;
#pragma unroll
  for (int j = 0; j < NT; ++j) {
    int o = tg * 64 + j * 16 + nn;
    float bias = pb[o];
#pragma unroll
    for (int r = 0; r < 8; ++r) {
      int m = tm * 16 + mb + r;                      // m = b*L + l
      int b = m >> 12, l = m & 4095;
      out[((size_t)b * INDIM + o) * LSEQ + l] = acc[j][r] + bias;
    }
  }
}

// ---------------------------------------------------------------------------
extern "C" void kernel_launch(void* const* d_in, const int* in_sizes, int n_in,
                              void* d_out, int out_size, void* d_ws, size_t ws_size,
                              hipStream_t stream) {
  const float* x    = (const float*)d_in[0];
  const float* g    = (const float*)d_in[1];
  const float* be   = (const float*)d_in[2];
  const float* Win  = (const float*)d_in[3];
  const float* cW   = (const float*)d_in[4];
  const float* cb   = (const float*)d_in[5];
  const float* Wxp  = (const float*)d_in[6];
  const float* dtW  = (const float*)d_in[7];
  const float* dtb  = (const float*)d_in[8];
  const float* Alog = (const float*)d_in[9];
  const float* Dp   = (const float*)d_in[10];
  const float* Wout = (const float*)d_in[11];
  const float* Wpj  = (const float*)d_in[12];
  const float* pb   = (const float*)d_in[13];
  const float* skip = (const float*)d_in[14];
  float* out = (float*)d_out;
  (void)in_sizes; (void)n_in; (void)out_size; (void)ws_size;

  char* ws = (char*)d_ws;
  size_t off = 0;
  auto take = [&](size_t bytes) -> char* {
    char* p = ws + off;
    off = (off + bytes + 255) & ~(size_t)255;
    return p;
  };
  const size_t MROWS = (size_t)NSEQ * LSEQ;          // 49152
  float*  xs   = (float*) take(sizeof(float)  * MROWS * DMODEL);
  __bf16* xsb  = (__bf16*)take(sizeof(__bf16) * MROWS * DMODEL);
  float*  xz   = (float*) take(sizeof(float)  * MROWS * 1024);
  float*  xc   = (float*) take(sizeof(float)  * MROWS * DINNER);
  __bf16* xcb  = (__bf16*)take(sizeof(__bf16) * MROWS * DINNER);
  float*  dbl  = (float*) take(sizeof(float)  * MROWS * XPPAD);
  float*  dtf  = (float*) take(sizeof(float)  * MROWS * DINNER);
  __bf16* yb   = (__bf16*)take(sizeof(__bf16) * MROWS * DINNER);
  float*  ycat = (float*) take(sizeof(float)  * (size_t)BATCH * LSEQ * INDIM);
  __bf16* ylnb = (__bf16*)take(sizeof(__bf16) * (size_t)BATCH * LSEQ * INDIM);
  __bf16* bWin = (__bf16*)take(sizeof(__bf16) * 1024 * 64);
  __bf16* bWxp = (__bf16*)take(sizeof(__bf16) * XPPAD * 512);
  __bf16* bWout= (__bf16*)take(sizeof(__bf16) * 64 * 512);
  __bf16* bWpj = (__bf16*)take(sizeof(__bf16) * 192 * 192);

  k_prep_weights<<<624, 256, 0, stream>>>(Win, Wxp, Wout, Wpj, bWin, bWxp, bWout, bWpj);
  k_ln1<<<64, 256, 0, stream>>>(x, g, be, xs, xsb);
  k_gemm_inproj<<<6144, 256, 0, stream>>>(xsb, bWin, xz);        // 49152 waves
  k_conv<<<98304, 256, 0, stream>>>(xz, cW, cb, xc, xcb);
  k_gemm_xproj<<<384, 256, 0, stream>>>(xcb, bWxp, dbl);         // 3072 waves
  k_dt<<<98304, 256, 0, stream>>>(dbl, dtW, dtb, dtf);
  k_scan<<<24, 256, 0, stream>>>(dbl, dtf, xc, xz, Alog, Dp, yb);
  k_gemm_outproj<<<384, 256, 0, stream>>>(yb, bWout, xs, skip, ycat);
  k_ln2<<<64, 256, 0, stream>>>(ycat, g, be, ylnb);
  k_gemm_proj<<<384, 256, 0, stream>>>(ylnb, bWpj, pb, out);     // 3072 waves
}
